// CausalAttention_5669356830735
// MI455X (gfx1250) — matrix operand order repaired and checked by reading, not compiled, not run on hardware
//
#include <hip/hip_runtime.h>

// ---------------------------------------------------------------------------
// Causal attention (B=4, T=2048, E=1024, D=128), fp32 in/out.
//   k0: W(fp32) -> bf16 once
//   k1: QKV projection (bf16 WMMA) + RoPE; Q,K row-major bf16, V transposed
//   k2: flash attention, v_wmma_f32_16x16x32_bf16, wave32-native softmax
// ---------------------------------------------------------------------------

typedef __attribute__((ext_vector_type(16))) __bf16 v16bf;
typedef __attribute__((ext_vector_type(8)))  __bf16 v8bf;
typedef __attribute__((ext_vector_type(8)))  float  v8f;

constexpr int   Bc    = 4;
constexpr int   Tc    = 2048;
constexpr int   Ec    = 1024;
constexpr int   Dc    = 128;
constexpr float SCALE = 0.08838834764831845f;   // 1/sqrt(128)
constexpr float THETA = 10000.0f;
constexpr int   WN    = Dc * Ec;                // elements per weight matrix

#define SHUF16(lo, hi) __builtin_shufflevector((lo), (hi), 0,1,2,3,4,5,6,7,8,9,10,11,12,13,14,15)

// ---------------------------------------------------------------------------
// Kernel 0: one-shot fp32 -> bf16 weight conversion (converted W lives in L2).
// ---------------------------------------------------------------------------
__global__ __launch_bounds__(256) void convert_w_kernel(
    const float* __restrict__ wQ, const float* __restrict__ wK,
    const float* __restrict__ wV,
    __bf16* __restrict__ q, __bf16* __restrict__ k, __bf16* __restrict__ v)
{
    int i = blockIdx.x * 256 + threadIdx.x;
    if (i < WN) {
        q[i] = (__bf16)wQ[i];
        k[i] = (__bf16)wK[i];
        v[i] = (__bf16)wV[i];
    }
}

// ---------------------------------------------------------------------------
// Kernel 1: Q/K/V projections (X[B*T,E] @ W^T) + RoPE on Q,K.
// One block = 16 rows, 256 threads = 8 waves; wave w owns output d-tile w.
// X tile staged in LDS as bf16 (32KB). Weights read as pure bf16 vectors.
// V written transposed [B][D][T] so the flash PV B-fragment is contiguous.
// ---------------------------------------------------------------------------
__global__ __launch_bounds__(256) void qkv_rope_kernel(
    const float*  __restrict__ X,
    const __bf16* __restrict__ wQb, const __bf16* __restrict__ wKb,
    const __bf16* __restrict__ wVb,
    __bf16* __restrict__ Qw, __bf16* __restrict__ Kw, __bf16* __restrict__ Vt)
{
    __shared__ __bf16 sX[16 * Ec];
    const int tid     = threadIdx.x;
    const int rowbase = blockIdx.x * 16;                 // over B*T (T%16==0)

    for (int i = tid; i < 16 * Ec; i += 256) {
        int r = i >> 10, c = i & (Ec - 1);
        sX[i] = (__bf16)X[(size_t)(rowbase + r) * Ec + c];
    }
    __syncthreads();

    const int ln   = tid & 31;
    const int wid  = tid >> 5;                           // 0..7
    const int g    = ln >> 4;
    const int lm   = ln & 15;
    const int dcol = wid * 16 + lm;                      // output column 0..127

    v8f cq = {}, ck = {}, cv = {};
    for (int k0 = 0; k0 < Ec / 32; ++k0) {
        // A-fragment: two contiguous 8-element runs in LDS (ds_load_b128 x2)
        const int ab = lm * Ec + k0 * 32 + 16 * g;
        v8bf alo = *(const v8bf*)&sX[ab];
        v8bf ahi = *(const v8bf*)&sX[ab + 8];
        v16bf a  = SHUF16(alo, ahi);
        // B-fragments: one contiguous 16-element run each (global b128 x2)
        const size_t wb = (size_t)dcol * Ec + k0 * 32 + 16 * g;
        v16bf bq = *(const v16bf*)&wQb[wb];
        v16bf bk = *(const v16bf*)&wKb[wb];
        v16bf bv = *(const v16bf*)&wVb[wb];
        cq = __builtin_amdgcn_wmma_f32_16x16x32_bf16(false, a, false, bq, (short)0, cq, false, false);
        ck = __builtin_amdgcn_wmma_f32_16x16x32_bf16(false, a, false, bk, (short)0, ck, false, false);
        cv = __builtin_amdgcn_wmma_f32_16x16x32_bf16(false, a, false, bv, (short)0, cv, false, false);
    }

    // RoPE: pair (even d, odd d) sits in adjacent lanes of the C layout.
    const float fr = __expf(-__logf(THETA) * (float)(dcol & ~1) / (float)Dc);
#pragma unroll
    for (int r = 0; r < 8; ++r) {
        int   row = rowbase + r + 8 * g;                 // C layout: M = r + 8g
        int   t   = row & (Tc - 1);                      // token position in batch
        int   b   = row >> 11;                           // row / T
        float ang = (float)t * fr;
        float cs  = __cosf(ang), sn = __sinf(ang);
        float xq  = cq[r],       xk = ck[r];
        float pq  = __shfl_xor(xq, 1, 32);
        float pk  = __shfl_xor(xk, 1, 32);
        float oq  = (dcol & 1) ? (pq * sn + xq * cs) : (xq * cs - pq * sn);
        float ok  = (dcol & 1) ? (pk * sn + xk * cs) : (xk * cs - pk * sn);
        Qw[(size_t)row * Dc + dcol] = (__bf16)oq;
        Kw[(size_t)row * Dc + dcol] = (__bf16)ok;
        Vt[((size_t)b * Dc + dcol) * Tc + t] = (__bf16)cv[r];   // transposed
    }
}

// ---------------------------------------------------------------------------
// Kernel 2: causal flash attention. One wave32 owns a 16-query tile; streams
// 32-key blocks: 8 WMMAs Q.K^T, online softmax, 8 WMMAs P.V.
// P transposed C-layout -> A-layout through bf16 LDS + s_wait_dscnt.
// ---------------------------------------------------------------------------
__global__ __launch_bounds__(128) void flash_attn_kernel(
    const __bf16* __restrict__ Qw, const __bf16* __restrict__ Kw,
    const __bf16* __restrict__ Vt, float* __restrict__ out)
{
    __shared__ __bf16 sPb[4][16 * 32];
    const int tid = threadIdx.x;
    const int ln  = tid & 31, w = tid >> 5;
    const int g   = ln >> 4,  lm = ln & 15;

    const int tile  = blockIdx.x * 4 + w;                // 0..511
    const int b     = tile / (Tc / 16);
    const int qbase = (tile % (Tc / 16)) * 16;           // query base in batch
    const size_t bofs = (size_t)b * Tc * Dc;             // also b*D*T for Vt

    // Q fragments for the 4 K-chunks of the D=128 contraction.
    v16bf qa[4];
#pragma unroll
    for (int c = 0; c < 4; ++c) {
        const size_t qb = bofs + (size_t)(qbase + lm) * Dc + c * 32 + 16 * g;
        v8bf lo = *(const v8bf*)&Qw[qb];
        v8bf hi = *(const v8bf*)&Qw[qb + 8];
        qa[c] = SHUF16(lo, hi);
    }

    v8f o[8] = {};
    float mi[8], li[8];
#pragma unroll
    for (int r = 0; r < 8; ++r) { mi[r] = -1e30f; li[r] = 0.f; }

    const int nblk = (qbase + 15) / 32 + 1;
    for (int kb = 0; kb < nblk; ++kb) {
        const int keybase = kb * 32;

        // Prefetch next key block's K rows and V^T lines (global_prefetch_b8)
        if (kb + 1 < nblk) {
            __builtin_prefetch(&Kw[bofs + (size_t)(keybase + 32 + ln) * Dc], 0, 0);
            __builtin_prefetch(&Vt[bofs + (size_t)(ln * 4) * Tc + keybase + 32], 0, 0);
        }

        // ---- scores S[16 x 32] = Q (16x128) . K^T -------------------------
        v8f s0 = {}, s1 = {};
#pragma unroll
        for (int c = 0; c < 4; ++c) {
            const size_t k0 = bofs + (size_t)(keybase + lm)      * Dc + c * 32 + 16 * g;
            const size_t k1 = bofs + (size_t)(keybase + 16 + lm) * Dc + c * 32 + 16 * g;
            v16bf b0 = *(const v16bf*)&Kw[k0];
            v16bf b1 = *(const v16bf*)&Kw[k1];
            s0 = __builtin_amdgcn_wmma_f32_16x16x32_bf16(false, qa[c], false, b0, (short)0, s0, false, false);
            s1 = __builtin_amdgcn_wmma_f32_16x16x32_bf16(false, qa[c], false, b1, (short)0, s1, false, false);
        }

        // ---- scale + causal mask -----------------------------------------
        const bool maskblk = (keybase + 31 > qbase);
#pragma unroll
        for (int r = 0; r < 8; ++r) {
            int   qi = qbase + r + 8 * g;
            float v0 = s0[r] * SCALE;
            float v1 = s1[r] * SCALE;
            if (maskblk) {
                if (keybase + lm      > qi) v0 = -1e30f;
                if (keybase + 16 + lm > qi) v1 = -1e30f;
            }
            s0[r] = v0; s1[r] = v1;
        }

        // ---- online softmax (row reductions across 16-lane groups) -------
        float p0[8], p1[8], corr[8];
#pragma unroll
        for (int r = 0; r < 8; ++r) {
            float bm = fmaxf(s0[r], s1[r]);
#pragma unroll
            for (int off = 1; off < 16; off <<= 1)
                bm = fmaxf(bm, __shfl_xor(bm, off, 32));
            float nm = fmaxf(mi[r], bm);
            corr[r] = __expf(mi[r] - nm);
            p0[r]   = __expf(s0[r] - nm);
            p1[r]   = __expf(s1[r] - nm);
            float rs = p0[r] + p1[r];
#pragma unroll
            for (int off = 1; off < 16; off <<= 1)
                rs += __shfl_xor(rs, off, 32);
            li[r] = li[r] * corr[r] + rs;
            mi[r] = nm;
        }
#pragma unroll
        for (int d = 0; d < 8; ++d)
#pragma unroll
            for (int r = 0; r < 8; ++r) o[d][r] *= corr[r];

        // ---- P: C-layout -> bf16 LDS -> A-layout fragment -----------------
#pragma unroll
        for (int r = 0; r < 8; ++r) {
            int m = r + 8 * g;
            sPb[w][m * 32 + lm]      = (__bf16)p0[r];
            sPb[w][m * 32 + 16 + lm] = (__bf16)p1[r];
        }
        asm volatile("s_wait_dscnt 0" ::: "memory");     // CDNA5 split DS counter
        v8bf plo = *(const v8bf*)&sPb[w][lm * 32 + 16 * g];
        v8bf phi = *(const v8bf*)&sPb[w][lm * 32 + 8 + 16 * g];
        v16bf pa = SHUF16(plo, phi);

        // ---- O += P (16x32) . V (32x128); V^T makes B-frag contiguous -----
#pragma unroll
        for (int d = 0; d < 8; ++d) {
            v16bf vb = *(const v16bf*)&Vt[bofs + (size_t)(d * 16 + lm) * Tc + keybase + 16 * g];
            o[d] = __builtin_amdgcn_wmma_f32_16x16x32_bf16(false, pa, false, vb, (short)0, o[d], false, false);
        }
    }

    // ---- finalize --------------------------------------------------------
#pragma unroll
    for (int d = 0; d < 8; ++d)
#pragma unroll
        for (int r = 0; r < 8; ++r) {
            int   qi  = qbase + r + 8 * g;
            float inv = (li[r] > 0.f) ? (1.f / li[r]) : 0.f;
            out[bofs + (size_t)qi * Dc + d * 16 + lm] = o[d][r] * inv;
        }
}

// ---------------------------------------------------------------------------
extern "C" void kernel_launch(void* const* d_in, const int* in_sizes, int n_in,
                              void* d_out, int out_size, void* d_ws, size_t ws_size,
                              hipStream_t stream)
{
    (void)in_sizes; (void)n_in; (void)out_size; (void)ws_size;
    const float* X  = (const float*)d_in[0];
    const float* wQ = (const float*)d_in[1];
    const float* wK = (const float*)d_in[2];
    const float* wV = (const float*)d_in[3];

    const size_t NTD = (size_t)Bc * Tc * Dc;             // 1M elements
    __bf16* Qw  = (__bf16*)d_ws;                         // bf16 scratch: 6MB + 0.75MB
    __bf16* Kw  = Qw  + NTD;
    __bf16* Vt  = Kw  + NTD;                             // [B][D][T]
    __bf16* wQb = Vt  + NTD;
    __bf16* wKb = wQb + WN;
    __bf16* wVb = wKb + WN;
    float*  out = (float*)d_out;

    convert_w_kernel<<<(WN + 255) / 256, 256, 0, stream>>>(wQ, wK, wV, wQb, wKb, wVb);
    qkv_rope_kernel<<<(Bc * Tc) / 16, 256, 0, stream>>>(X, wQb, wKb, wVb, Qw, Kw, Vt);
    flash_attn_kernel<<<(Bc * (Tc / 16)) / 4, 128, 0, stream>>>(Qw, Kw, Vt, out);
}
